// ChannelAttention_71004399337608
// MI455X (gfx1250) — matrix operand order repaired and checked by reading, not compiled, not run on hardware
//
#include <hip/hip_runtime.h>
#include <hip/hip_bf16.h>
#include <math.h>

// ---------------------------------------------------------------------------
// Problem constants
// ---------------------------------------------------------------------------
#define Bdim   8
#define Cdim   192
#define C3dim  576
#define Hdim   128
#define Wdim   128
#define Ndim   16384          // H*W
#define HEADS  6
#define CH     32             // C / HEADS
#define BCN    (Bdim*Cdim*Ndim)   // 25,165,824

typedef __attribute__((ext_vector_type(16))) __bf16 bf16x16;
typedef __attribute__((ext_vector_type(8)))  float  f32x8;

__device__ __forceinline__ unsigned f2bf(float f) {
    unsigned u = __float_as_uint(f);
    return (u + 0x7FFFu + ((u >> 16) & 1u)) >> 16;   // RNE to bf16
}
__device__ __forceinline__ float bf2f(unsigned short h) {
    return __uint_as_float(((unsigned)h) << 16);
}
__device__ __forceinline__ float sigmoidf_(float x) { return 1.f / (1.f + expf(-x)); }
__device__ __forceinline__ float gelu_exact(float x) {
    return 0.5f * x * (1.f + erff(x * 0.70710678118654752f));
}

// ---------------------------------------------------------------------------
// K0: weight conversion fp32 -> bf16
// ---------------------------------------------------------------------------
__global__ void cvt_weights(const float* __restrict__ wq, const float* __restrict__ wp,
                            unsigned short* __restrict__ wqb, unsigned short* __restrict__ wpb) {
    int i = blockIdx.x * 256 + threadIdx.x;
    if (i < C3dim * Cdim) wqb[i] = (unsigned short)f2bf(wq[i]);
    if (i < Cdim * Cdim)  wpb[i] = (unsigned short)f2bf(wp[i]);
}

// ---------------------------------------------------------------------------
// WMMA GEMM  Y[b,r,n] = sum_k W[r,k] * X[b,k,n],  Cin = 192.
// Block = 128 threads (4 waves). Block tile: 64 rows x 64 cols.
// Wave w owns the 16-row stripe row0 + 16w; all 4 waves share one B panel,
// and each wave reuses its A fragment across 4 column tiles (24 WMMA/wave).
// LDS holds packed bf16 k-pairs so each fragment element is one dword read
// matching the CDNA5 16-bit A(16x32)/B(32x16) VGPR layouts:
//   A: lane<16 -> K 0..7,16..23 halves; lane>=16 -> +8. VGPR j: K pair.
//   B: same K order as A (lane<16 -> K 0..7,16..23; lane>=16 -> +8).
// ---------------------------------------------------------------------------
template <bool X_BF16, bool OUT_BF16>
__global__ void gemm192_wmma(const unsigned short* __restrict__ Wb, // [R,192] bf16
                             const void* __restrict__ Xv,           // [B,192,N]
                             void* __restrict__ Yv,                 // [B,R,N]
                             int R) {
    __shared__ unsigned As2[96 * 64];   // [k_pair][row]  (64 rows)
    __shared__ unsigned Bs2[96 * 64];   // [k_pair][col]  (64 cols)
    const int tid  = threadIdx.x;
    const int b    = blockIdx.z;
    const int row0 = blockIdx.y * 64;
    const int n0   = blockIdx.x * 64;

    // ---- stage A (weights), 64 rows x 192 k, as k-pairs
    for (int i = tid; i < 96 * 64; i += 128) {
        int row = i & 63, pr = i >> 6;
        const unsigned short* wp = Wb + (row0 + row) * Cdim + 2 * pr;
        As2[pr * 64 + row] = (unsigned)wp[0] | ((unsigned)wp[1] << 16);
    }
    // ---- stage B (activations), 192 k x 64 n, as k-pairs
    if (X_BF16) {
        const unsigned short* X = (const unsigned short*)Xv;
        for (int i = tid; i < 96 * 64; i += 128) {
            int col = i & 63, pr = i >> 6;
            long base = (long)(b * Cdim + 2 * pr) * Ndim + n0 + col;
            Bs2[pr * 64 + col] = (unsigned)X[base] | ((unsigned)X[base + Ndim] << 16);
        }
    } else {
        const float* X = (const float*)Xv;
        for (int i = tid; i < 96 * 64; i += 128) {
            int col = i & 63, pr = i >> 6;
            long base = (long)(b * Cdim + 2 * pr) * Ndim + n0 + col;
            Bs2[pr * 64 + col] = f2bf(X[base]) | (f2bf(X[base + Ndim]) << 16);
        }
    }
    __syncthreads();

    const int wave = tid >> 5, lane = tid & 31;
    const int half = lane >> 4, lr = lane & 15;
    const int wrow = wave * 16;          // this wave's row stripe within block

    union { bf16x16 v; unsigned u[8]; } a, bm;
    f32x8 acc[4];
#pragma unroll
    for (int ct = 0; ct < 4; ++ct) acc[ct] = (f32x8){0.f,0.f,0.f,0.f,0.f,0.f,0.f,0.f};

#pragma unroll
    for (int kk = 0; kk < 6; ++kk) {           // K = 6 * 32 = 192
        // A fragment for this wave's 16 rows (reused across 4 col tiles)
#pragma unroll
        for (int j = 0; j < 8; ++j) {
            int koffA = (j < 4 ? 2 * j : 16 + 2 * (j - 4)) + half * 8;
            a.u[j] = As2[((kk * 32 + koffA) >> 1) * 64 + wrow + lr];
        }
#pragma unroll
        for (int ct = 0; ct < 4; ++ct) {
#pragma unroll
            for (int j = 0; j < 8; ++j) {
                int koffB = (j < 4 ? 2 * j : 16 + 2 * (j - 4)) + half * 8;
                bm.u[j] = Bs2[((kk * 32 + koffB) >> 1) * 64 + ct * 16 + lr];
            }
            acc[ct] = __builtin_amdgcn_wmma_f32_16x16x32_bf16(
                false, a.v, false, bm.v, (short)0, acc[ct], false, false);
        }
    }

    // ---- store: D layout lane<16 -> M=v, lane>=16 -> M=v+8; col = lr
    long ybase = (long)(b * R + row0 + wrow) * Ndim + n0 + lr;
    if (OUT_BF16) {
        unsigned short* Y = (unsigned short*)Yv;
#pragma unroll
        for (int ct = 0; ct < 4; ++ct)
#pragma unroll
            for (int v2 = 0; v2 < 8; ++v2) {
                int m = v2 + half * 8;
                Y[ybase + (long)m * Ndim + ct * 16] = (unsigned short)f2bf(acc[ct][v2]);
            }
    } else {
        float* Y = (float*)Yv;
#pragma unroll
        for (int ct = 0; ct < 4; ++ct)
#pragma unroll
            for (int v2 = 0; v2 < 8; ++v2) {
                int m = v2 + half * 8;
                Y[ybase + (long)m * Ndim + ct * 16] = acc[ct][v2];
            }
    }
}

// ---------------------------------------------------------------------------
// K2: depthwise 3x3 (pad 1, no bias) on bf16 qkv -> fp32 q/k/v buffers
// ---------------------------------------------------------------------------
__global__ void dwconv_qkv(const unsigned short* __restrict__ qkv,
                           const float* __restrict__ wdw,  // [576,1,3,3]
                           float* __restrict__ qb, float* __restrict__ kb,
                           float* __restrict__ vb) {
    long idx = (long)blockIdx.x * 256 + threadIdx.x;       // B*576*N
    int n  = (int)(idx & (Ndim - 1));
    int c3 = (int)((idx >> 14) % C3dim);
    int b  = (int)(idx / ((long)Ndim * C3dim));
    int x = n & (Wdim - 1), y = n >> 7;
    const unsigned short* src = qkv + ((long)(b * C3dim + c3) << 14);
    const float* wp = wdw + c3 * 9;
    float s = 0.f;
#pragma unroll
    for (int dy = -1; dy <= 1; ++dy) {
        int yy = y + dy; if (yy < 0 || yy > Hdim - 1) continue;
#pragma unroll
        for (int dx = -1; dx <= 1; ++dx) {
            int xx = x + dx; if (xx < 0 || xx > Wdim - 1) continue;
            s += bf2f(src[yy * Wdim + xx]) * wp[(dy + 1) * 3 + dx + 1];
        }
    }
    int g = c3 / Cdim, cc = c3 % Cdim;
    float* dst = (g == 0) ? qb : ((g == 1) ? kb : vb);
    dst[((long)(b * Cdim + cc) << 14) + n] = s;
}

// ---------------------------------------------------------------------------
// K3: per-row inverse L2 norm over N (eps-clamped like F.normalize)
// ---------------------------------------------------------------------------
__global__ void rownorm(const float* __restrict__ q, const float* __restrict__ k,
                        float* __restrict__ qinv, float* __restrict__ kinv) {
    int r = blockIdx.x;                               // 0 .. 2*1536-1
    int rr = r & 1535;
    const float* src = ((r < 1536) ? q : k) + ((long)rr << 14);
    float ss = 0.f;
    for (int i = threadIdx.x; i < Ndim; i += 256) { float v = src[i]; ss += v * v; }
    __shared__ float red[256];
    red[threadIdx.x] = ss; __syncthreads();
    for (int s2 = 128; s2 > 0; s2 >>= 1) {
        if (threadIdx.x < s2) red[threadIdx.x] += red[threadIdx.x + s2];
        __syncthreads();
    }
    if (threadIdx.x == 0) {
        float inv = 1.f / fmaxf(sqrtf(red[0]), 1e-12f);
        ((r < 1536) ? qinv : kinv)[rr] = inv;
    }
}

// ---------------------------------------------------------------------------
// K4: attn[b,h,c,d] = softmax_d( scale * qhat_c . khat_d ), K reduction = 16384
// One block per (b,h); LDS-tiled fp32 dot products, then row softmax.
// ---------------------------------------------------------------------------
__global__ void attn_softmax_k(const float* __restrict__ q, const float* __restrict__ k,
                               const float* __restrict__ qinv, const float* __restrict__ kinv,
                               float* __restrict__ attn) {
    int bh = blockIdx.x, b = bh / HEADS, h = bh % HEADS;
    const float* qs = q + ((long)(b * Cdim + h * CH) << 14);
    const float* ks = k + ((long)(b * Cdim + h * CH) << 14);
    __shared__ float qt[32][65];
    __shared__ float kt[32][65];
    __shared__ float S[32][33];
    int tid = threadIdx.x;
    float acc[4] = {0.f, 0.f, 0.f, 0.f};
    for (int n0 = 0; n0 < Ndim; n0 += 64) {
        for (int i = tid; i < 2048; i += 256) {
            int row = i >> 6, col = i & 63;
            qt[row][col] = qs[((long)row << 14) + n0 + col];
            kt[row][col] = ks[((long)row << 14) + n0 + col];
        }
        __syncthreads();
#pragma unroll
        for (int p = 0; p < 4; ++p) {
            int pair = tid + p * 256;           // 0..1023
            int c = pair >> 5, d = pair & 31;
            float s = 0.f;
#pragma unroll 8
            for (int t = 0; t < 64; ++t) s += qt[c][t] * kt[d][t];
            acc[p] += s;
        }
        __syncthreads();
    }
    const float scale = 0.17677669529663687f;   // 1/sqrt(32)
#pragma unroll
    for (int p = 0; p < 4; ++p) {
        int pair = tid + p * 256;
        int c = pair >> 5, d = pair & 31;
        S[c][d] = acc[p] * qinv[b * Cdim + h * CH + c] * kinv[b * Cdim + h * CH + d] * scale;
    }
    __syncthreads();
    if (tid < 32) {
        float mx = -1e30f;
        for (int d = 0; d < 32; ++d) mx = fmaxf(mx, S[tid][d]);
        float sum = 0.f;
        for (int d = 0; d < 32; ++d) { float e = expf(S[tid][d] - mx); S[tid][d] = e; sum += e; }
        float inv = 1.f / sum;
        for (int d = 0; d < 32; ++d) attn[(bh * 32 + tid) * 32 + d] = S[tid][d] * inv;
    }
}

// ---------------------------------------------------------------------------
// K5: out[b,h,c,n] = sum_d attn[c,d] * v[d,n]   (LDS-tiled, 64 n per block)
// ---------------------------------------------------------------------------
__global__ void attn_apply(const float* __restrict__ attn, const float* __restrict__ v,
                           float* __restrict__ out) {
    int b = blockIdx.z, h = blockIdx.y, n0 = blockIdx.x * 64;
    __shared__ float At[32][33];
    __shared__ float Vt[32][65];
    int tid = threadIdx.x;
    for (int i = tid; i < 1024; i += 256)
        At[i >> 5][i & 31] = attn[((b * HEADS + h) * 32 + (i >> 5)) * 32 + (i & 31)];
    const float* vs = v + ((long)(b * Cdim + h * CH) << 14) + n0;
    for (int i = tid; i < 2048; i += 256)
        Vt[i >> 6][i & 63] = vs[((long)(i >> 6) << 14) + (i & 63)];
    __syncthreads();
    int col = tid & 63, c0 = (tid >> 6) * 8;
    float* op = out + ((long)(b * Cdim + h * CH + c0) << 14) + n0 + col;
#pragma unroll
    for (int ci = 0; ci < 8; ++ci) {
        float s = 0.f;
#pragma unroll
        for (int d = 0; d < 32; ++d) s += At[c0 + ci][d] * Vt[d][col];
        op[(long)ci << 14] = s;
    }
}

// ---------------------------------------------------------------------------
// K6: depthwise 3x3 on v + bias -> cx  (fp32, coalesced)
// ---------------------------------------------------------------------------
__global__ void dwconv_v(const float* __restrict__ v, const float* __restrict__ w,
                         const float* __restrict__ bias, float* __restrict__ cx) {
    long idx = (long)blockIdx.x * 256 + threadIdx.x;   // B*C*N
    int n = (int)(idx & (Ndim - 1));
    long bc = idx >> 14;
    int c = (int)(bc % Cdim);
    int x = n & (Wdim - 1), y = n >> 7;
    const float* src = v + (bc << 14);
    const float* wp = w + c * 9;
    float s = bias[c];
#pragma unroll
    for (int dy = -1; dy <= 1; ++dy) {
        int yy = y + dy; if (yy < 0 || yy > Hdim - 1) continue;
#pragma unroll
        for (int dx = -1; dx <= 1; ++dx) {
            int xx = x + dx; if (xx < 0 || xx > Wdim - 1) continue;
            s += src[yy * Wdim + xx] * wp[(dy + 1) * 3 + dx + 1];
        }
    }
    cx[idx] = s;
}

// ---------------------------------------------------------------------------
// K7: channel LayerNorm (over C) + affine + exact GELU, in place on cx.
// One thread per (b, n) column -> fully coalesced across n.
// ---------------------------------------------------------------------------
__global__ void ln_gelu(float* __restrict__ cx, const float* __restrict__ lnw,
                        const float* __restrict__ lnb) {
    int idx = blockIdx.x * 256 + threadIdx.x;          // B*N
    int b = idx >> 14, n = idx & (Ndim - 1);
    float* col = cx + (((long)b * Cdim) << 14) + n;
    float s = 0.f, s2 = 0.f;
    for (int c = 0; c < Cdim; ++c) { float v = col[(long)c << 14]; s += v; s2 += v * v; }
    float mean = s * (1.f / Cdim);
    float var = s2 * (1.f / Cdim) - mean * mean;
    float inv = rsqrtf(var + 1e-5f);
    for (int c = 0; c < Cdim; ++c) {
        float v = (col[(long)c << 14] - mean) * inv * lnw[c] + lnb[c];
        col[(long)c << 14] = gelu_exact(v);
    }
}

// ---------------------------------------------------------------------------
// K8: global average pool of out over N  ->  pooled[b*C + c]
// ---------------------------------------------------------------------------
__global__ void rowmean(const float* __restrict__ out, float* __restrict__ pooled) {
    int r = blockIdx.x;                                // B*C rows
    const float* src = out + ((long)r << 14);
    float s = 0.f;
    for (int i = threadIdx.x; i < Ndim; i += 256) s += src[i];
    __shared__ float red[256];
    red[threadIdx.x] = s; __syncthreads();
    for (int s2 = 128; s2 > 0; s2 >>= 1) {
        if (threadIdx.x < s2) red[threadIdx.x] += red[threadIdx.x + s2];
        __syncthreads();
    }
    if (threadIdx.x == 0) pooled[r] = red[0] * (1.f / Ndim);
}

// ---------------------------------------------------------------------------
// K9: SE-style channel MLP: gelu(pooled@W1^T+b1)@W2^T+b2 -> sigmoid gate
// ---------------------------------------------------------------------------
__global__ void channel_mlp(const float* __restrict__ pooled,
                            const float* __restrict__ w1, const float* __restrict__ b1,
                            const float* __restrict__ w2, const float* __restrict__ b2,
                            float* __restrict__ gate) {
    int b = blockIdx.x, t = threadIdx.x;
    __shared__ float p[Cdim];
    __shared__ float hid[24];
    if (t < Cdim) p[t] = pooled[b * Cdim + t];
    __syncthreads();
    if (t < 24) {
        float s = b1[t];
        for (int i = 0; i < Cdim; ++i) s += w1[t * Cdim + i] * p[i];
        hid[t] = gelu_exact(s);
    }
    __syncthreads();
    if (t < Cdim) {
        float s = b2[t];
        for (int j = 0; j < 24; ++j) s += w2[t * 24 + j] * hid[j];
        gate[b * Cdim + t] = sigmoidf_(s);
    }
}

// ---------------------------------------------------------------------------
// K10: per-pixel avg & max over C of conv_x  -> sm [2, B*N]
// ---------------------------------------------------------------------------
__global__ void spatial_stats(const float* __restrict__ cx, float* __restrict__ sm) {
    int idx = blockIdx.x * 256 + threadIdx.x;          // B*N
    int b = idx >> 14, n = idx & (Ndim - 1);
    const float* src = cx + (((long)b * Cdim) << 14) + n;
    float s = 0.f, m = -3.0e38f;
    for (int c = 0; c < Cdim; ++c) { float v = src[(long)c << 14]; s += v; m = fmaxf(m, v); }
    sm[idx] = s * (1.f / Cdim);
    sm[Bdim * Ndim + idx] = m;
}

// ---------------------------------------------------------------------------
// K11: 7x7 conv (pad 3) over [avg;max] + bias -> sigmoid spatial gate [B*N]
// ---------------------------------------------------------------------------
__global__ void spatial_conv(const float* __restrict__ sm, const float* __restrict__ wsi,
                             const float* __restrict__ bsi, float* __restrict__ gate) {
    int idx = blockIdx.x * 256 + threadIdx.x;
    int b = idx >> 14, n = idx & (Ndim - 1);
    int x = n & (Wdim - 1), y = n >> 7;
    float s = bsi[0];
    for (int p2 = 0; p2 < 2; ++p2) {
        const float* plane = sm + p2 * (Bdim * Ndim) + ((long)b << 14);
        const float* wp = wsi + p2 * 49;
        for (int dy = -3; dy <= 3; ++dy) {
            int yy = y + dy; if (yy < 0 || yy > Hdim - 1) continue;
            for (int dx = -3; dx <= 3; ++dx) {
                int xx = x + dx; if (xx < 0 || xx > Wdim - 1) continue;
                s += plane[yy * Wdim + xx] * wp[(dy + 3) * 7 + dx + 3];
            }
        }
    }
    gate[idx] = sigmoidf_(s);
}

// ---------------------------------------------------------------------------
// K12: result = out*sig(spatial) + conv_x*sig(channel), store bf16 for GEMM
// ---------------------------------------------------------------------------
__global__ void combine(const float* __restrict__ out, const float* __restrict__ cx,
                        const float* __restrict__ spg, const float* __restrict__ chg,
                        unsigned short* __restrict__ res) {
    long idx = (long)blockIdx.x * 256 + threadIdx.x;   // B*C*N
    int n = (int)(idx & (Ndim - 1));
    long bc = idx >> 14;
    int c = (int)(bc % Cdim);
    int b = (int)(bc / Cdim);
    float r = out[idx] * spg[(b << 14) + n] + cx[idx] * chg[b * Cdim + c];
    res[idx] = (unsigned short)f2bf(r);
}

// ---------------------------------------------------------------------------
// Host-side launcher
// ---------------------------------------------------------------------------
extern "C" void kernel_launch(void* const* d_in, const int* in_sizes, int n_in,
                              void* d_out, int out_size, void* d_ws, size_t ws_size,
                              hipStream_t stream) {
    const float* x      = (const float*)d_in[0];
    const float* w_qkv  = (const float*)d_in[1];
    const float* w_dw   = (const float*)d_in[2];
    const float* w_dcv  = (const float*)d_in[3];
    const float* b_dcv  = (const float*)d_in[4];
    const float* ln_w   = (const float*)d_in[5];
    const float* ln_b   = (const float*)d_in[6];
    const float* w_ci1  = (const float*)d_in[7];
    const float* b_ci1  = (const float*)d_in[8];
    const float* w_ci2  = (const float*)d_in[9];
    const float* b_ci2  = (const float*)d_in[10];
    const float* w_si   = (const float*)d_in[11];
    const float* b_si   = (const float*)d_in[12];
    const float* w_proj = (const float*)d_in[13];

    char* W = (char*)d_ws;
    const size_t bcn = (size_t)BCN;
    // region A: bf16 qkv (pre-depthwise); later reused for bf16 result
    unsigned short* qkv_b = (unsigned short*)W;                       // 3*bcn*2 B
    // region B: fp32 q,k,v (post-depthwise); q->out, k->conv_x reuse
    float* qbuf = (float*)(W + 3 * bcn * 2);
    float* kbuf = qbuf + bcn;
    float* vbuf = kbuf + bcn;
    float* outb = qbuf;          // q dead after attn
    float* cxb  = kbuf;          // k dead after attn
    unsigned short* resb = qkv_b; // qkv dead after depthwise
    // region C: small buffers
    char* S = W + 3 * bcn * 2 + 3 * bcn * 4;
    unsigned short* wqb = (unsigned short*)S;  S += (size_t)C3dim * Cdim * 2;
    unsigned short* wpb = (unsigned short*)S;  S += (size_t)Cdim * Cdim * 2;
    float* qinv  = (float*)S;  S += 1536 * 4;
    float* kinv  = (float*)S;  S += 1536 * 4;
    float* attnb = (float*)S;  S += (size_t)Bdim * HEADS * 32 * 32 * 4;
    float* pooled= (float*)S;  S += 1536 * 4;
    float* chg   = (float*)S;  S += 1536 * 4;
    float* smb   = (float*)S;  S += (size_t)2 * Bdim * Ndim * 4;
    float* spg   = (float*)S;  S += (size_t)Bdim * Ndim * 4;

    // 0) weights -> bf16
    cvt_weights<<<432, 256, 0, stream>>>(w_qkv, w_proj, wqb, wpb);
    // 1) qkv 1x1 conv as WMMA GEMM (bf16 out), 64x64 block tiles
    gemm192_wmma<false, true><<<dim3(Ndim / 64, C3dim / 64, Bdim), 128, 0, stream>>>(
        wqb, (const void*)x, (void*)qkv_b, C3dim);
    // 2) depthwise 3x3 on qkv -> q,k,v
    dwconv_qkv<<<(Bdim * C3dim * Ndim) / 256, 256, 0, stream>>>(qkv_b, w_dw, qbuf, kbuf, vbuf);
    // 3) row norms
    rownorm<<<2 * Bdim * Cdim, 256, 0, stream>>>(qbuf, kbuf, qinv, kinv);
    // 4) attention + softmax
    attn_softmax_k<<<Bdim * HEADS, 256, 0, stream>>>(qbuf, kbuf, qinv, kinv, attnb);
    // 5) out = attn @ v   (overwrites q region)
    attn_apply<<<dim3(Ndim / 64, HEADS, Bdim), 256, 0, stream>>>(attnb, vbuf, outb);
    // 6) conv branch: depthwise 3x3 + bias on v -> cx (k region)
    dwconv_v<<<(Bdim * Cdim * Ndim) / 256, 256, 0, stream>>>(vbuf, w_dcv, b_dcv, cxb);
    // 7) channel LN + GELU in place
    ln_gelu<<<(Bdim * Ndim) / 256, 256, 0, stream>>>(cxb, ln_w, ln_b);
    // 8) GAP(out)
    rowmean<<<Bdim * Cdim, 256, 0, stream>>>(outb, pooled);
    // 9) channel gate
    channel_mlp<<<Bdim, 256, 0, stream>>>(pooled, w_ci1, b_ci1, w_ci2, b_ci2, chg);
    // 10) spatial stats (avg/max over C)
    spatial_stats<<<(Bdim * Ndim) / 256, 256, 0, stream>>>(cxb, smb);
    // 11) 7x7 conv + sigmoid spatial gate
    spatial_conv<<<(Bdim * Ndim) / 256, 256, 0, stream>>>(smb, w_si, b_si, spg);
    // 12) cross gating + residual combine -> bf16 result
    combine<<<(Bdim * Cdim * Ndim) / 256, 256, 0, stream>>>(outb, cxb, spg, chg, resb);
    // 13) output 1x1 projection as WMMA GEMM (fp32 out -> d_out)
    gemm192_wmma<true, false><<<dim3(Ndim / 64, Cdim / 64, Bdim), 128, 0, stream>>>(
        wpb, (const void*)resb, d_out, Cdim);
}
